// Attention_80839874445487
// MI455X (gfx1250) — matrix operand order repaired and checked by reading, not compiled, not run on hardware
//
#include <hip/hip_runtime.h>
#include <hip/hip_bf16.h>

// ---------------- problem constants ----------------
#define BATCH   8
#define SEQ     1024
#define DIM     768
#define HEADS   12
#define HD      64            // head dim
#define QSCALE  0.125f        // 64^-0.5
#define NEGINF  -65504.0f
#define QKV_ELEMS ((size_t)BATCH * HEADS * SEQ * HD)   // 6291456

typedef __attribute__((ext_vector_type(16))) _Float16 v16h;
typedef __attribute__((ext_vector_type(8)))  _Float16 v8h;
typedef __attribute__((ext_vector_type(8)))  float    v8f;

__device__ __forceinline__ v16h cat8(v8h lo, v8h hi) {
    return __builtin_shufflevector(lo, hi, 0,1,2,3,4,5,6,7,8,9,10,11,12,13,14,15);
}
__device__ __forceinline__ v8f wmma_f16(v16h a, v16h b, v8f c) {
    // D = A(16x32 f16) * B(32x16 f16) + C(16x16 f32)
    return __builtin_amdgcn_wmma_f32_16x16x32_f16(false, a, false, b, (short)0, c, false, false);
}

// ---- optional CDNA5 async global->LDS copy (ASYNCcnt path) ----
#if defined(__has_builtin)
#if __has_builtin(__builtin_amdgcn_global_load_async_to_lds_b128)
#define HAVE_ASYNC_LDS 1
#endif
#endif

#ifdef HAVE_ASYNC_LDS
// Param 1 (from hipcc diagnostic): GCC-vector int4 pointer in AS(1).
typedef int v4i_g __attribute__((vector_size(16)));
typedef __attribute__((address_space(1))) v4i_g* glob_v4i;
typedef __attribute__((address_space(3))) v4i_g* lds_v4i;

__device__ __forceinline__ void async_copy16B(const void* g, void* l) {
    __builtin_amdgcn_global_load_async_to_lds_b128(
        (glob_v4i)(g), (lds_v4i)(l), 0, 0);
}
__device__ __forceinline__ void async_drain() {
    asm volatile("s_wait_asynccnt 0" ::: "memory");
}
#endif

// =====================================================================
// Generic 64x64-tile GEMM:  D[M x NOUT] = A[M x K] * W[NOUT x K]^T + bias
// MODE 0: A = f32 (x); epilogue scatters f16 into q/k (row-major [B,H,N,64],
//         q pre-scaled) and v TRANSPOSED ([B,H,64,N]) so the attention
//         kernel's P@V B-fragments are K-contiguous global loads.
// MODE 1: A = f16 (attn output), epilogue writes f32 to d_out.
// Block: 128 threads = 4 waves. Wave w owns rows [16w,16w+16) of the tile,
// all 64 cols (4 accumulators). K staged in 32-wide LDS chunks as f16.
// =====================================================================
#define LDK 40   // padded K-stride in halves (80B: 16B aligned, conflict-free)

template<int MODE>
__global__ __launch_bounds__(128)
void gemm64x64(const float* __restrict__ Af32, const _Float16* __restrict__ Ah16,
               const float* __restrict__ Wf32, const float* __restrict__ bias,
               _Float16* __restrict__ outH, float* __restrict__ outF,
               int M, int K, int NOUT)
{
    __shared__ _Float16 sA[64 * LDK];
    __shared__ _Float16 sB[64 * LDK];

    const int t    = threadIdx.x;
    const int wave = t >> 5;
    const int lane = t & 31;
    const int m16  = lane & 15;   // row/col within 16
    const int hh   = lane >> 4;   // lane-half select
    const int rowBase = blockIdx.y * 64;
    const int colBase = blockIdx.x * 64;

    // staging coords: 2 threads per row, 16 consecutive K each
    const int lr = t >> 1;          // 0..63
    const int lk = (t & 1) * 16;    // 0 or 16

    v8f acc[4] = {};

    for (int k0 = 0; k0 < K; k0 += 32) {
        __syncthreads();
        // ---- stage A tile (64 rows x 32 K) as f16 ----
        if constexpr (MODE == 0) {
            const float* src = Af32 + (size_t)(rowBase + lr) * K + (k0 + lk);
            #pragma unroll
            for (int i = 0; i < 16; ++i)
                sA[lr * LDK + lk + i] = (_Float16)src[i];
        } else {
            const _Float16* src = Ah16 + (size_t)(rowBase + lr) * K + (k0 + lk);
#ifdef HAVE_ASYNC_LDS
            async_copy16B(src,     &sA[lr * LDK + lk]);       // 8 halves
            async_copy16B(src + 8, &sA[lr * LDK + lk + 8]);   // 8 halves
#else
            *(v8h*)&sA[lr * LDK + lk]     = *(const v8h*)src;
            *(v8h*)&sA[lr * LDK + lk + 8] = *(const v8h*)(src + 8);
#endif
        }
        // ---- stage W tile (64 out-cols x 32 K) as f16 ----
        {
            const float* src = Wf32 + (size_t)(colBase + lr) * K + (k0 + lk);
            #pragma unroll
            for (int i = 0; i < 16; ++i)
                sB[lr * LDK + lk + i] = (_Float16)src[i];
        }
        if (k0 + 32 < K) {  // prefetch next W chunk (global_prefetch_b8)
            __builtin_prefetch((const char*)(Wf32 + (size_t)(colBase + lr) * K + (k0 + 32 + lk)), 0, 1);
        }
#ifdef HAVE_ASYNC_LDS
        if constexpr (MODE == 1) async_drain();
#endif
        __syncthreads();

        // ---- A fragment: lane holds row (wave*16+m16); K per ISA layout ----
        const int am = wave * 16 + m16;
        v8h a0 = *(const v8h*)&sA[am * LDK + 8 * hh];        // K = 8*hh + 0..7
        v8h a1 = *(const v8h*)&sA[am * LDK + 16 + 8 * hh];   // K = 16 + 8*hh + 0..7
        v16h afrag = cat8(a0, a1);

        #pragma unroll
        for (int j = 0; j < 4; ++j) {
            // B fragment: lane holds out-col (j*16+m16); K = 8*hh + 0..7, 16 + 8*hh + 0..7
            const int bn = j * 16 + m16;
            v8h b0 = *(const v8h*)&sB[bn * LDK + 8 * hh];
            v8h b1 = *(const v8h*)&sB[bn * LDK + 16 + 8 * hh];
            acc[j] = wmma_f16(afrag, cat8(b0, b1), acc[j]);
        }
    }

    // ---- epilogue: lane holds D[wave*16 + 8*hh + r][j*16 + m16] ----
    const int orow0 = rowBase + wave * 16 + 8 * hh;
    #pragma unroll
    for (int j = 0; j < 4; ++j) {
        const int gc = colBase + j * 16 + m16;
        const float bv = bias[gc];
        #pragma unroll
        for (int r = 0; r < 8; ++r) {
            const int gm = orow0 + r;
            float val = acc[j][r] + bv;
            if constexpr (MODE == 0) {
                const int which = gc / DIM;      // 0=q 1=k 2=v
                const int c  = gc % DIM;
                const int h  = c / HD, d = c % HD;
                const int bb = gm / SEQ, nn = gm % SEQ;
                const size_t bhI = (size_t)(bb * HEADS + h);
                if (which == 2) {
                    // v stored transposed: [B,H,64,N]
                    outH[2 * QKV_ELEMS + (bhI * HD + d) * SEQ + nn] = (_Float16)val;
                } else {
                    if (which == 0) val *= QSCALE;   // fold softmax scale into q
                    outH[(size_t)which * QKV_ELEMS + (bhI * SEQ + nn) * HD + d] = (_Float16)val;
                }
            } else {
                outF[(size_t)gm * NOUT + gc] = val;
            }
        }
    }
}

// =====================================================================
// Attention core: one block per (b, h, 16-row tile).
// Phase 1: S = q*k^T (WMMA, K=64), + mask/bias -> d_out attn + LDS
// Phase 2: row softmax over 1024 cols (f32 in LDS), P -> f16 in LDS
// Phase 3: O = P*V (WMMA, K=1024 in 32-chunks); V is pre-transposed
//          [B,H,64,N] so B-fragments are direct 16B global loads.
// =====================================================================
#define SS_STRIDE 1028                 // f32, padded
#define SP_STRIDE 1032                 // f16, padded (16B-aligned rows)
#define SS_BYTES  (16 * SS_STRIDE * 4)             // 65792
#define SP_BYTES  (16 * SP_STRIDE * 2)             // 33024
#define SMEM_BYTES (SS_BYTES + SP_BYTES + 16*8*4 + 16*4 + 16*4)

__global__ __launch_bounds__(128)
void attn_kernel(const _Float16* __restrict__ qS, const _Float16* __restrict__ kH,
                 const _Float16* __restrict__ vT, const unsigned char* __restrict__ mask,
                 const float* __restrict__ bias, float* __restrict__ attnOut,
                 _Float16* __restrict__ attO)
{
    extern __shared__ char smemRaw[];
    float*    sS   = (float*)smemRaw;                         // [16][1028]
    _Float16* sP   = (_Float16*)(smemRaw + SS_BYTES);         // [16][1032]
    float*    sRed = (float*)(smemRaw + SS_BYTES + SP_BYTES); // [16][8]
    float*    sMax = sRed + 16 * 8;                           // [16]
    float*    sSum = sMax + 16;                               // [16]

    const int t    = threadIdx.x;
    const int wave = t >> 5;
    const int lane = t & 31;
    const int m16  = lane & 15;
    const int hh   = lane >> 4;

    const int bh      = blockIdx.x >> 6;         // b*HEADS + h
    const int rowTile = blockIdx.x & 63;
    const int b       = bh / HEADS;
    const int h       = bh % HEADS;
    const int rowBase = rowTile * 16;

    const _Float16* qrow  = qS + (((size_t)bh) * SEQ + rowBase) * HD + (size_t)m16 * HD;
    const _Float16* kbase = kH + ((size_t)bh) * SEQ * HD;
    const _Float16* vbase = vT + ((size_t)bh) * HD * SEQ;   // [64][1024]

    // ---------- Phase 1: scores ----------
    v16h aq[2];
    #pragma unroll
    for (int ks = 0; ks < 2; ++ks) {
        v8h x0 = *(const v8h*)(qrow + ks * 32 + 8 * hh);
        v8h x1 = *(const v8h*)(qrow + ks * 32 + 16 + 8 * hh);
        aq[ks] = cat8(x0, x1);
    }
    const size_t maskBase = (size_t)b * SEQ * SEQ;
    const size_t biasBase = (size_t)h * SEQ * SEQ;
    const size_t attnBase = (size_t)bh * SEQ * SEQ;

    for (int ct = wave; ct < 64; ct += 4) {           // 16 col-tiles per wave
        const int cb = ct * 16;
        const _Float16* krow = kbase + (size_t)(cb + m16) * HD;
        v8f acc = {};
        #pragma unroll
        for (int ks = 0; ks < 2; ++ks) {
            v8h b0 = *(const v8h*)(krow + ks * 32 + 8 * hh);
            v8h b1 = *(const v8h*)(krow + ks * 32 + 16 + 8 * hh);
            acc = wmma_f16(aq[ks], cat8(b0, b1), acc);
        }
        const int col = cb + m16;
        #pragma unroll
        for (int r = 0; r < 8; ++r) {
            const int row  = rowBase + 8 * hh + r;
            const int rloc = 8 * hh + r;
            float s = acc[r];
            if (mask[maskBase + (size_t)row * SEQ + col]) s = NEGINF;
            s += bias[biasBase + (size_t)row * SEQ + col];
            attnOut[attnBase + (size_t)row * SEQ + col] = s;   // attn_wo_softmax
            sS[rloc * SS_STRIDE + col] = s;
        }
    }
    __syncthreads();

    // ---------- Phase 2: softmax (8 threads per row, interleaved cols) ----------
    const int sr = t >> 3;   // row 0..15
    const int sc = t & 7;    // sub-lane 0..7
    float mx = -3.0e38f;
    for (int i = 0; i < 128; ++i) mx = fmaxf(mx, sS[sr * SS_STRIDE + sc + 8 * i]);
    sRed[sr * 8 + sc] = mx;
    __syncthreads();
    if (sc == 0) {
        float m2 = sRed[sr * 8];
        #pragma unroll
        for (int i = 1; i < 8; ++i) m2 = fmaxf(m2, sRed[sr * 8 + i]);
        sMax[sr] = m2;
    }
    __syncthreads();
    const float rm = sMax[sr];
    float sum = 0.f;
    for (int i = 0; i < 128; ++i) {
        const int c = sc + 8 * i;
        const float e = __expf(sS[sr * SS_STRIDE + c] - rm);
        sS[sr * SS_STRIDE + c] = e;
        sum += e;
    }
    sRed[sr * 8 + sc] = sum;
    __syncthreads();
    if (sc == 0) {
        float s2 = 0.f;
        #pragma unroll
        for (int i = 0; i < 8; ++i) s2 += sRed[sr * 8 + i];
        sSum[sr] = s2;
    }
    __syncthreads();
    const float inv = 1.0f / sSum[sr];
    for (int i = 0; i < 128; ++i) {
        const int c = sc + 8 * i;
        sP[sr * SP_STRIDE + c] = (_Float16)(sS[sr * SS_STRIDE + c] * inv);
    }
    __syncthreads();   // sP ready for all waves

    // ---------- Phase 3: O = P @ V (V pre-transposed; no LDS staging) ----------
    const int dcol = wave * 16 + m16;                 // output head-dim column
    const _Float16* vrow = vbase + (size_t)dcol * SEQ;  // K-contiguous
    v8f accO = {};
    for (int kc = 0; kc < 32; ++kc) {
        v8h a0 = *(const v8h*)&sP[m16 * SP_STRIDE + kc * 32 + 8 * hh];
        v8h a1 = *(const v8h*)&sP[m16 * SP_STRIDE + kc * 32 + 16 + 8 * hh];
        v8h b0 = *(const v8h*)(vrow + kc * 32 + 8 * hh);
        v8h b1 = *(const v8h*)(vrow + kc * 32 + 16 + 8 * hh);
        accO = wmma_f16(cat8(a0, a1), cat8(b0, b1), accO);
    }
    // attO[b][row][h*64 + dcol] (f16, feeds proj GEMM)
    #pragma unroll
    for (int r = 0; r < 8; ++r) {
        const int row = rowBase + 8 * hh + r;
        attO[((size_t)b * SEQ + row) * DIM + h * HD + dcol] = (_Float16)accO[r];
    }
}

// =====================================================================
extern "C" void kernel_launch(void* const* d_in, const int* in_sizes, int n_in,
                              void* d_out, int out_size, void* d_ws, size_t ws_size,
                              hipStream_t stream)
{
    const float*         x     = (const float*)d_in[0];
    const float*         rpb   = (const float*)d_in[1];           // [12,1024,1024]
    const unsigned char* mask  = (const unsigned char*)d_in[2];   // bool [8,1024,1024]
    const float*         Wqkv  = (const float*)d_in[3];           // [2304,768]
    const float*         bqkv  = (const float*)d_in[4];
    const float*         Wproj = (const float*)d_in[5];           // [768,768]
    const float*         bproj = (const float*)d_in[6];

    float* out     = (float*)d_out;                               // [8,1024,768]
    float* attnOut = out + (size_t)BATCH * SEQ * DIM;             // [8,12,1024,1024]

    _Float16* qS   = (_Float16*)d_ws;          // q (pre-scaled), [B,H,N,64] f16
    _Float16* kH   = qS + QKV_ELEMS;           // [B,H,N,64] f16
    _Float16* vTr  = kH + QKV_ELEMS;           // v transposed [B,H,64,N] f16
    _Float16* attO = vTr + QKV_ELEMS;          // [B,N,768] f16
    (void)in_sizes; (void)n_in; (void)out_size; (void)ws_size;

    // K1: qkv = x @ Wqkv^T + b  (M=8192, K=768, NOUT=2304)
    {
        dim3 grid(3 * DIM / 64, BATCH * SEQ / 64);   // 36 x 128
        gemm64x64<0><<<grid, 128, 0, stream>>>(x, nullptr, Wqkv, bqkv,
                                               qS, nullptr, BATCH * SEQ, DIM, 3 * DIM);
    }
    // K2: attention core + attn_wo_softmax output
    {
        const int grid = BATCH * HEADS * (SEQ / 16); // 6144
        attn_kernel<<<grid, 128, SMEM_BYTES, stream>>>(qS, kH, vTr, mask, rpb,
                                                       attnOut, attO);
    }
    // K3: out = attO @ Wproj^T + b  (M=8192, K=768, NOUT=768)
    {
        dim3 grid(DIM / 64, BATCH * SEQ / 64);       // 12 x 128
        gemm64x64<1><<<grid, 128, 0, stream>>>(nullptr, attO, Wproj, bproj,
                                               nullptr, out, BATCH * SEQ, DIM, DIM);
    }
}
